// SpikeMLP_2800318677740
// MI455X (gfx1250) — compile-verified, matrix operand order checked by reading, not run on hardware
//
#include <hip/hip_runtime.h>
#include <hip/hip_bf16.h>

typedef __attribute__((ext_vector_type(16))) _Float16 v16h;
typedef __attribute__((ext_vector_type(8)))  _Float16 v8h;
typedef __attribute__((ext_vector_type(4)))  _Float16 v4h;
typedef __attribute__((ext_vector_type(8)))  float    v8f;
typedef __attribute__((ext_vector_type(4)))  float    v4f;

#define B_SZ    2048
#define D_IN    1024
#define T_SZ    64
#define H1      512
#define H2      512
#define D_OUT   512
#define M_TOT   (B_SZ * T_SZ)   // 131072

// ---------------------------------------------------------------------------
// A/B fragment loader for V_WMMA_F32_16X16X32_F16.
// Row-major, K-contiguous source (rows = M for A, rows = N for B since we
// compute A @ W.T with W stored [N,K] row-major -> identical addressing).
// Per ISA 7.12.2: lane<16 holds K{0..7,16..23}, lane>=16 holds K{8..15,24..31}.
// ---------------------------------------------------------------------------
__device__ inline v16h load_frag(const _Float16* __restrict__ base, int ld, int lane) {
    const int r = lane & 15;
    const int h = lane >> 4;
    const _Float16* p = base + (size_t)r * ld + h * 8;
    v8h lo = *(const v8h*)p;          // K = h*8 .. h*8+7
    v8h hi = *(const v8h*)(p + 16);   // K = 16 + h*8 .. 16 + h*8 + 7
    return __builtin_shufflevector(lo, hi, 0,1,2,3,4,5,6,7,8,9,10,11,12,13,14,15);
}

__device__ inline v8f wmma_f16(v16h a, v16h b, v8f c) {
    return __builtin_amdgcn_wmma_f32_16x16x32_f16(
        false, a, false, b, (short)0, c, false, false);
}

// ---------------------------------------------------------------------------
// f32 -> f16 weight conversion
// ---------------------------------------------------------------------------
__global__ void cvt_f16_kernel(const float* __restrict__ in,
                               _Float16* __restrict__ out, int n) {
    int i = blockIdx.x * 256 + threadIdx.x;
    if (i < n) out[i] = (_Float16)in[i];
}

// ---------------------------------------------------------------------------
// Transpose spikes [B, D_IN, T] f32 -> Xh[(t*B + b), D_IN] f16
// One block: one b, 64 d's. Coalesced along T on load, along D on store.
// ---------------------------------------------------------------------------
__global__ __launch_bounds__(256) void transpose_kernel(
    const float* __restrict__ in, _Float16* __restrict__ Xh) {
    __shared__ float tile[64][65];
    const int b  = blockIdx.y;
    const int d0 = blockIdx.x * 64;
    const int tid = threadIdx.x;
    for (int i = 0; i < 16; ++i) {
        int idx = i * 256 + tid;
        int dl = idx >> 6, tt = idx & 63;
        tile[dl][tt] = in[((size_t)b * D_IN + d0 + dl) * T_SZ + tt];
    }
    __syncthreads();
    for (int i = 0; i < 16; ++i) {
        int idx = i * 256 + tid;
        int tt = idx >> 6, dl = idx & 63;
        Xh[((size_t)(tt * B_SZ + b)) * D_IN + d0 + dl] = (_Float16)tile[dl][tt];
    }
}

// ---------------------------------------------------------------------------
// GEMM1: C1h[m, n] = Xh[m, :] @ W1h[n, :] + b1[n], m = t*B+b.
// 8 waves/block (4 m-waves x 2 n-waves). Wave tile 32x64 = 2x4 WMMA accums.
// Grid: (N/128 = 4, M/128 = 1024); x-fastest order keeps A panels hot in L2.
// ---------------------------------------------------------------------------
__global__ __launch_bounds__(256) void gemm1_kernel(
    const _Float16* __restrict__ Xh, const _Float16* __restrict__ W1h,
    const float* __restrict__ b1, _Float16* __restrict__ C1h) {
    const int lane = threadIdx.x & 31;
    const int wave = threadIdx.x >> 5;
    const int wm = wave & 3, wn = wave >> 2;
    const int m0 = blockIdx.y * 128 + wm * 32;
    const int n0 = blockIdx.x * 128 + wn * 64;
    const int l15 = lane & 15;

    v8f acc[2][4];
    for (int jn = 0; jn < 4; ++jn) {
        float bv = b1[n0 + jn * 16 + l15];
        for (int im = 0; im < 2; ++im)
            for (int e = 0; e < 8; ++e) acc[im][jn][e] = bv;
    }
    for (int k = 0; k < D_IN; k += 32) {
        v16h a0 = load_frag(Xh + (size_t)m0 * D_IN + k, D_IN, lane);
        v16h a1 = load_frag(Xh + (size_t)(m0 + 16) * D_IN + k, D_IN, lane);
        v16h bf[4];
        for (int jn = 0; jn < 4; ++jn)
            bf[jn] = load_frag(W1h + (size_t)(n0 + jn * 16) * D_IN + k, D_IN, lane);
        for (int jn = 0; jn < 4; ++jn) {
            acc[0][jn] = wmma_f16(a0, bf[jn], acc[0][jn]);
            acc[1][jn] = wmma_f16(a1, bf[jn], acc[1][jn]);
        }
    }
    const int rh = (lane >> 4) * 8;   // C layout: row = rh + e, col = l15
    for (int im = 0; im < 2; ++im)
        for (int jn = 0; jn < 4; ++jn) {
            int col = n0 + jn * 16 + l15;
            for (int e = 0; e < 8; ++e)
                C1h[(size_t)(m0 + im * 16 + rh + e) * H1 + col] =
                    (_Float16)acc[im][jn][e];
        }
}

// ---------------------------------------------------------------------------
// Fused: layer-1 scan -> GEMM2 -> layer-2 scan -> acc, then GEMM3 epilogue.
// Block = 16 batch rows x full H1/H2. 8 waves, each owns 64 columns
// (4 WMMA tiles) of H2/D_OUT. Neuron state w = v*(1-s)*0.75 lives in regs.
// s1 tile shared through LDS as an f16 A-matrix (16 x 512).
// ---------------------------------------------------------------------------
__global__ __launch_bounds__(256) void fused_snn_kernel(
    const _Float16* __restrict__ C1h, const _Float16* __restrict__ W2h,
    const float* __restrict__ b2, const _Float16* __restrict__ Wouth,
    const float* __restrict__ bout, float* __restrict__ out) {
    __shared__ _Float16 sA[16 * H1];   // 16 KB: s1 tile, reused for acc tile
    const int tid  = threadIdx.x;
    const int lane = tid & 31;
    const int wave = tid >> 5;
    const int l15  = lane & 15;
    const int rh   = (lane >> 4) * 8;
    const int b0   = blockIdx.x * 16;

    // layer-1 state: 32 of the 16x512 elements per thread (coalesced float4s)
    float w1s[8][4];
    for (int g = 0; g < 8; ++g)
        for (int e = 0; e < 4; ++e) w1s[g][e] = 0.0f;

    v8f w2s[4], accum[4];
    float b2w[4], bow[4];
    for (int jn = 0; jn < 4; ++jn) {
        int n = wave * 64 + jn * 16 + l15;
        b2w[jn] = b2[n];
        bow[jn] = bout[n];
        for (int e = 0; e < 8; ++e) { w2s[jn][e] = 0.0f; accum[jn][e] = 0.0f; }
    }

    for (int t = 0; t < T_SZ; ++t) {
        // ---- layer-1 scan: v1 = w1 + c1; s1 = v1>0.5; w1 = v1*(1-s1)*0.75
        for (int g = 0; g < 8; ++g) {
            int flat = (g * 256 + tid) * 4;
            int r = flat >> 9, col = flat & 511;
            v4h c1v = *(const v4h*)(C1h +
                ((size_t)(t * B_SZ + b0 + r) << 9) + col);
            v4h sh;
            for (int e = 0; e < 4; ++e) {
                float v = w1s[g][e] + (float)c1v[e];
                float s = v > 0.5f ? 1.0f : 0.0f;
                w1s[g][e] = v * (1.0f - s) * 0.75f;
                sh[e] = (_Float16)s;
            }
            *(v4h*)(sA + flat) = sh;
        }
        __syncthreads();

        // ---- GEMM2: c2[16x64] = s1[16x512] @ W2[cols].T + b2
        v8f c2[4];
        for (int jn = 0; jn < 4; ++jn)
            for (int e = 0; e < 8; ++e) c2[jn][e] = b2w[jn];
        for (int k = 0; k < H1; k += 32) {
            v16h a = load_frag(sA + k, H1, lane);   // ds_load from LDS
            for (int jn = 0; jn < 4; ++jn) {
                v16h bf = load_frag(
                    W2h + (size_t)(wave * 64 + jn * 16) * H1 + k, H1, lane);
                c2[jn] = wmma_f16(a, bf, c2[jn]);
            }
        }
        // ---- layer-2 scan in accumulator layout
        for (int jn = 0; jn < 4; ++jn)
            for (int e = 0; e < 8; ++e) {
                float v = w2s[jn][e] + c2[jn][e];
                float s = v > 0.5f ? 1.0f : 0.0f;
                w2s[jn][e] = v * (1.0f - s) * 0.75f;
                accum[jn][e] += s;
            }
        __syncthreads();   // protect sA before next t overwrites it
    }

    // ---- acc tile -> LDS as f16 A-matrix (counts <= 64: exact in f16)
    for (int jn = 0; jn < 4; ++jn) {
        int col = wave * 64 + jn * 16 + l15;
        for (int e = 0; e < 8; ++e)
            sA[(rh + e) * H1 + col] = (_Float16)accum[jn][e];
    }
    __syncthreads();

    // ---- GEMM3: out = acc @ Wout.T + bout
    v8f o[4];
    for (int jn = 0; jn < 4; ++jn)
        for (int e = 0; e < 8; ++e) o[jn][e] = bow[jn];
    for (int k = 0; k < H2; k += 32) {
        v16h a = load_frag(sA + k, H2, lane);
        for (int jn = 0; jn < 4; ++jn) {
            v16h bf = load_frag(
                Wouth + (size_t)(wave * 64 + jn * 16) * H2 + k, H2, lane);
            o[jn] = wmma_f16(a, bf, o[jn]);
        }
    }
    for (int jn = 0; jn < 4; ++jn) {
        int col = wave * 64 + jn * 16 + l15;
        for (int e = 0; e < 8; ++e)
            out[(size_t)(b0 + rh + e) * D_OUT + col] = o[jn][e];
    }
}

// ---------------------------------------------------------------------------
extern "C" void kernel_launch(void* const* d_in, const int* in_sizes, int n_in,
                              void* d_out, int out_size, void* d_ws, size_t ws_size,
                              hipStream_t stream) {
    (void)in_sizes; (void)n_in; (void)out_size; (void)ws_size;
    const float* spikes = (const float*)d_in[0];   // [2048,1024,64]
    const float* W1     = (const float*)d_in[1];   // [512,1024]
    const float* b1     = (const float*)d_in[2];
    const float* W2     = (const float*)d_in[3];   // [512,512]
    const float* b2     = (const float*)d_in[4];
    const float* Wout   = (const float*)d_in[5];   // [512,512]
    const float* bout   = (const float*)d_in[6];
    float* out = (float*)d_out;

    char* ws = (char*)d_ws;
    size_t off = 0;
    _Float16* Xh    = (_Float16*)(ws + off); off += (size_t)M_TOT * D_IN * 2;  // 256 MB
    _Float16* C1h   = (_Float16*)(ws + off); off += (size_t)M_TOT * H1   * 2;  // 128 MB
    _Float16* W1h   = (_Float16*)(ws + off); off += (size_t)H1 * D_IN * 2;
    _Float16* W2h   = (_Float16*)(ws + off); off += (size_t)H2 * H1   * 2;
    _Float16* Wouth = (_Float16*)(ws + off); off += (size_t)D_OUT * H2 * 2;

    cvt_f16_kernel<<<(H1 * D_IN + 255) / 256, 256, 0, stream>>>(W1, W1h, H1 * D_IN);
    cvt_f16_kernel<<<(H2 * H1 + 255) / 256, 256, 0, stream>>>(W2, W2h, H2 * H1);
    cvt_f16_kernel<<<(D_OUT * H2 + 255) / 256, 256, 0, stream>>>(Wout, Wouth, D_OUT * H2);

    transpose_kernel<<<dim3(D_IN / 64, B_SZ), 256, 0, stream>>>(spikes, Xh);

    gemm1_kernel<<<dim3(H1 / 128, M_TOT / 128), 256, 0, stream>>>(Xh, W1h, b1, C1h);

    fused_snn_kernel<<<B_SZ / 16, 256, 0, stream>>>(C1h, W2h, b2, Wouth, bout, out);
}
